// Ssm_11570641895774
// MI455X (gfx1250) — compile-verified, matrix operand order checked by reading, not run on hardware
//
#include <hip/hip_runtime.h>
#include <hip/hip_bf16.h>

// ---------------------------------------------------------------------------
// Types for CDNA5 WMMA / TDM
// ---------------------------------------------------------------------------
typedef __attribute__((ext_vector_type(16))) _Float16 v16h;
typedef __attribute__((ext_vector_type(8)))  float    v8f;
typedef __attribute__((ext_vector_type(4)))  unsigned int u32x4;
typedef __attribute__((ext_vector_type(4)))  int      i32x4;
typedef __attribute__((ext_vector_type(8)))  int      i32x8;

#define BM 128
#define BN 128
#define BK 64
#define IMGS 64          // B_ * L_ = 4*16
#define OHW 14           // spatial at the big-conv stage
#define MPOS (IMGS * OHW * OHW)   // 12544
#define NOUT 512

__device__ __forceinline__ float gelu_erf(float x) {
    return 0.5f * x * (1.0f + erff(x * 0.70710678118654752f));
}

// ---------------------------------------------------------------------------
// Direct conv (small channel counts) + folded BN + optional residual/ReLU.
// ---------------------------------------------------------------------------
__global__ void conv3_bn_kernel(const float* __restrict__ x, const float* __restrict__ w,
                                const float* __restrict__ g, const float* __restrict__ b,
                                const float* __restrict__ mu, const float* __restrict__ var,
                                const float* __restrict__ res, float* __restrict__ out,
                                int Cin, int Co, int Hin, int Win, int Ho, int Wo,
                                int stride, int pad, int ks, int relu, int total)
{
    int i = blockIdx.x * blockDim.x + threadIdx.x;
    if (i >= total) return;
    int ow = i % Wo;  int t1 = i / Wo;
    int oh = t1 % Ho; int t2 = t1 / Ho;
    int co = t2 % Co; int img = t2 / Co;

    float acc = 0.0f;
    for (int ci = 0; ci < Cin; ++ci) {
        for (int kh = 0; kh < ks; ++kh) {
            int ih = oh * stride + kh - pad;
            if (ih < 0 || ih >= Hin) continue;
            for (int kw = 0; kw < ks; ++kw) {
                int iw = ow * stride + kw - pad;
                if (iw < 0 || iw >= Win) continue;
                acc += x[((size_t)(img * Cin + ci) * Hin + ih) * Win + iw] *
                       w[((size_t)(co * Cin + ci) * ks + kh) * ks + kw];
            }
        }
    }
    float inv = g[co] * rsqrtf(var[co] + 1e-5f);
    float val = acc * inv + (b[co] - mu[co] * inv);
    if (res)  val += res[i];
    if (relu) val = fmaxf(val, 0.0f);
    out[i] = val;
}

// ---------------------------------------------------------------------------
// Repack OIHW f32 weights -> [N][Kpad] f16 (K = ci*ks*ks ordering, zero pad).
// ---------------------------------------------------------------------------
__global__ void pack_w_f16(const float* __restrict__ w, _Float16* __restrict__ wp,
                           int Cin, int ks, int Kpad, int Co)
{
    int i = blockIdx.x * blockDim.x + threadIdx.x;
    int total = Co * Kpad;
    if (i >= total) return;
    int k = i % Kpad;
    int n = i / Kpad;
    int KK = ks * ks;
    int Kreal = Cin * KK;
    _Float16 v = (_Float16)0.0f;
    if (k < Kreal) {
        int ci = k / KK;
        int r  = k - ci * KK;
        int kh = r / ks;
        int kw = r - kh * ks;
        v = (_Float16)w[((size_t)(n * Cin + ci) * ks + kh) * ks + kw];
    }
    wp[(size_t)n * Kpad + k] = v;
}

// ---------------------------------------------------------------------------
// Implicit-GEMM conv with WMMA (f16 inputs, f32 accum), templated on CIN/KS so
// all coordinate decode uses compile-time divisors.
//   A[M,K] : fused im2col of x (f32 NCHW 14x14, stride 1, SAME pad)
//   B[K,N] : packed f16 weights [N][Kpad] -> streamed by the Tensor Data Mover
//   out    : f32 NCHW [IMGS,512,14,14], epilogue = BN (+res) (+ReLU)
// 256 threads = 8 waves (2x4); tile 128x128x64; 16 v_wmma per K-stage.
// ---------------------------------------------------------------------------
template<int CIN, int KS>
__global__ void __launch_bounds__(256)
conv_igemm_wmma(const float* __restrict__ x, const _Float16* __restrict__ wp,
                float* __restrict__ out, const float* __restrict__ res,
                const float* __restrict__ bng, const float* __restrict__ bnb,
                const float* __restrict__ bnm, const float* __restrict__ bnv,
                int Kpad, int relu)
{
    constexpr int KK    = KS * KS;
    constexpr int Kreal = CIN * KK;
    constexpr int PAD   = (KS == 3) ? 1 : 0;

    __shared__ __align__(16) _Float16 As[BM][BK];
    __shared__ __align__(16) _Float16 Bs[BN][BK];

    const int t    = threadIdx.x;
    const int lane = t & 31;
    const int wv   = t >> 5;      // wave 0..7
    const int wm   = wv >> 2;     // 0..1 : 64-row slab
    const int wn   = wv & 3;      // 0..3 : 32-col slab
    const int hi   = lane >> 4;   // half-wave select
    const int l15  = lane & 15;

    const int m0 = blockIdx.x * BM;
    const int n0 = blockIdx.y * BN;

    // Each thread owns one A row (mi = t>>1) and a 32-wide K slice.
    const int mi    = t >> 1;
    const int kbase = (t & 1) * 32;
    const int m     = m0 + mi;
    const int img   = m / (OHW * OHW);            // compile-time divisor
    const int rem   = m - img * (OHW * OHW);
    const int oh    = rem / OHW;
    const int ow    = rem - oh * OHW;
    const float* ximg = x + (size_t)img * CIN * (OHW * OHW);

    v8f acc[4][2];
    for (int i = 0; i < 4; ++i)
        for (int j = 0; j < 2; ++j)
            acc[i][j] = v8f{};

    union AU { v16h v; unsigned int d[8]; };

    for (int k0 = 0; k0 < Kpad; k0 += BK) {
        // ---- B tile: one Tensor-DMA per workgroup (wave 0 issues) ----
#if __has_builtin(__builtin_amdgcn_tensor_load_to_lds)
        if (t == 0) {
            unsigned ldsa = (unsigned)(size_t)(&Bs[0][0]);
            unsigned long long ga =
                (unsigned long long)(size_t)(wp + (size_t)n0 * Kpad + k0);
            union { u32x4 v; unsigned a[4]; } g0;
            g0.a[0] = 1u;                                          // count=1
            g0.a[1] = ldsa;                                        // lds_addr
            g0.a[2] = (unsigned)(ga & 0xFFFFFFFFu);                // gaddr lo
            g0.a[3] = (unsigned)((ga >> 32) & 0x1FFFFFFu) | (2u << 30); // hi + type=2
            union { i32x8 v; unsigned a[8]; } g1;
            g1.a[0] = (1u << 16);                                  // data_size=2B
            g1.a[1] = ((unsigned)Kpad & 0xFFFFu) << 16;            // tdim0 lo16 @48
            g1.a[2] = ((unsigned)Kpad >> 16) | (512u << 16);       // tdim0 hi, tdim1 lo
            g1.a[3] = ((unsigned)BK << 16);                        // tdim1 hi=0, tile0=BK
            g1.a[4] = (unsigned)BN;                                // tile1=BN, tile2=0
            g1.a[5] = (unsigned)Kpad;                              // tdim0_stride lo32
            g1.a[6] = 0u;
            g1.a[7] = 0u;
            i32x4 z4 = {0, 0, 0, 0};
#if defined(__clang_major__) && (__clang_major__ >= 23)
            i32x8 z8 = {0, 0, 0, 0, 0, 0, 0, 0};
            __builtin_amdgcn_tensor_load_to_lds(g0.v, g1.v, z4, z4, z8, 0);
#else
            __builtin_amdgcn_tensor_load_to_lds(g0.v, g1.v, z4, z4, 0);
#endif
        }
#else
        // fallback: plain staged copy
        for (int e = 0; e < 32; ++e)
            Bs[mi][kbase + e] = wp[(size_t)(n0 + mi) * Kpad + (k0 + kbase + e)];
#endif

        // ---- A tile: fused im2col f32->f16 (overlaps the TDM transfer) ----
        {
            int k  = k0 + kbase;
            int ci = k / KK;                    // compile-time divisor
            int r  = k - ci * KK;
            int kh = (KS == 1) ? 0 : (r / KS);
            int kw = (KS == 1) ? 0 : (r - kh * KS);
            for (int e = 0; e < 32; ++e) {
                float val = 0.0f;
                if (ci < CIN) {
                    int ih = oh + kh - PAD;
                    int iw = ow + kw - PAD;
                    if (ih >= 0 && ih < OHW && iw >= 0 && iw < OHW)
                        val = ximg[ci * (OHW * OHW) + ih * OHW + iw];
                }
                As[mi][kbase + e] = (_Float16)val;
                // incremental (kw,kh,ci) advance -- no division
                if (KS == 1) {
                    ++ci;
                } else {
                    if (++kw == KS) { kw = 0; if (++kh == KS) { kh = 0; ++ci; } }
                }
            }
        }

#if __has_builtin(__builtin_amdgcn_tensor_load_to_lds)
        __builtin_amdgcn_s_wait_tensorcnt(0);
#endif
        __syncthreads();

        // ---- 2 K-substeps x 8 tiles = 16 v_wmma per stage ----
        for (int kt = 0; kt < 2; ++kt) {
            AU a[4], bfr[2];
            for (int mt = 0; mt < 4; ++mt) {
                const unsigned int* ap =
                    (const unsigned int*)(&As[wm * 64 + mt * 16 + l15][kt * 32]);
                for (int q = 0; q < 8; ++q) {
                    int kk = 2 * q + ((q < 4) ? 0 : 8) + hi * 8;   // even K index
                    a[mt].d[q] = ap[kk >> 1];
                }
            }
            for (int nt = 0; nt < 2; ++nt) {
                const unsigned int* bp =
                    (const unsigned int*)(&Bs[wn * 32 + nt * 16 + l15][kt * 32]);
                for (int q = 0; q < 8; ++q) {
                    int kk = 2 * q + ((q < 4) ? 0 : 8) + hi * 8;
                    bfr[nt].d[q] = bp[kk >> 1];
                }
            }
            for (int mt = 0; mt < 4; ++mt)
                for (int nt = 0; nt < 2; ++nt)
                    acc[mt][nt] = __builtin_amdgcn_wmma_f32_16x16x32_f16(
                        false, a[mt].v, false, bfr[nt].v,
                        (short)0, acc[mt][nt], false, false);
        }
        __syncthreads();
    }

    // ---- epilogue: BN affine (+residual) (+ReLU), scatter to NCHW ----
    for (int mt = 0; mt < 4; ++mt) {
        for (int nt = 0; nt < 2; ++nt) {
            int n = n0 + wn * 32 + nt * 16 + l15;
            float inv = bng[n] * rsqrtf(bnv[n] + 1e-5f);
            float sh  = bnb[n] - bnm[n] * inv;
            for (int r = 0; r < 8; ++r) {
                int me  = m0 + wm * 64 + mt * 16 + hi * 8 + r;
                int im2 = me / (OHW * OHW);
                int rm2 = me - im2 * (OHW * OHW);
                int oh2 = rm2 / OHW;
                int ow2 = rm2 - oh2 * OHW;
                size_t o = ((size_t)(im2 * NOUT + n) * OHW + oh2) * OHW + ow2;
                float val = acc[mt][nt][r] * inv + sh;
                if (res)  val += res[o];
                if (relu) val = fmaxf(val, 0.0f);
                out[o] = val;
            }
        }
    }
}

// ---------------------------------------------------------------------------
// Global average pool over 14x14 : [64,512,14,14] -> [64,512]
// ---------------------------------------------------------------------------
__global__ void pool_kernel(const float* __restrict__ x, float* __restrict__ out)
{
    int i = blockIdx.x * blockDim.x + threadIdx.x;
    if (i >= IMGS * NOUT) return;
    const float* p = x + (size_t)i * (OHW * OHW);
    float s = 0.0f;
    for (int j = 0; j < OHW * OHW; ++j) s += p[j];
    out[i] = s * (1.0f / (OHW * OHW));
}

// ---------------------------------------------------------------------------
// LayerNorm over last dim D per row; one 256-thread block per row.
// ---------------------------------------------------------------------------
__global__ void ln_kernel(const float* __restrict__ x, const float* __restrict__ g,
                          const float* __restrict__ b, float* __restrict__ y, int D)
{
    __shared__ float red[256];
    int row = blockIdx.x;
    int t   = threadIdx.x;
    const float* xr = x + (size_t)row * D;

    float s = 0.0f;
    for (int j = t; j < D; j += 256) s += xr[j];
    red[t] = s; __syncthreads();
    for (int o = 128; o > 0; o >>= 1) { if (t < o) red[t] += red[t + o]; __syncthreads(); }
    float mu = red[0] / (float)D;
    __syncthreads();

    float s2 = 0.0f;
    for (int j = t; j < D; j += 256) { float d = xr[j] - mu; s2 += d * d; }
    red[t] = s2; __syncthreads();
    for (int o = 128; o > 0; o >>= 1) { if (t < o) red[t] += red[t + o]; __syncthreads(); }
    float inv = rsqrtf(red[0] / (float)D + 1e-5f);

    for (int j = t; j < D; j += 256)
        y[(size_t)row * D + j] = (xr[j] - mu) * inv * g[j] + b[j];
}

// ---------------------------------------------------------------------------
// Tiny GEMM: C[m,n] = act( dot(A[m,:], W[n,:]) + bias[n] + res[m,n] ); W [N,K]
// ---------------------------------------------------------------------------
__global__ void gemm_small(const float* __restrict__ A, const float* __restrict__ W,
                           const float* __restrict__ bias, const float* __restrict__ res,
                           float* __restrict__ C, int M, int N, int K, int relu)
{
    int i = blockIdx.x * blockDim.x + threadIdx.x;
    if (i >= M * N) return;
    int n = i % N, m = i / N;
    const float* a = A + (size_t)m * K;
    const float* w = W + (size_t)n * K;
    float s = 0.0f;
    for (int k = 0; k < K; ++k) s += a[k] * w[k];
    if (bias) s += bias[n];
    if (res)  s += res[i];
    if (relu) s = fmaxf(s, 0.0f);
    C[i] = s;
}

// ---------------------------------------------------------------------------
// S5: ZOH coefficients  Lb = exp(lam*dt),  coef = (Lb-1)/lam   (complex)
// ---------------------------------------------------------------------------
__global__ void s5_coeff(const float* __restrict__ lam_re, const float* __restrict__ lam_im,
                         const float* __restrict__ log_step,
                         float* Lbr, float* Lbi, float* cor, float* coi)
{
    int p = blockIdx.x * blockDim.x + threadIdx.x;
    if (p >= 512) return;
    float dt = expf(log_step[p]);
    float ar = lam_re[p] * dt, ai = lam_im[p] * dt;
    float er = expf(ar);
    float lr = er * cosf(ai), li = er * sinf(ai);
    Lbr[p] = lr; Lbi[p] = li;
    float nr = lr - 1.0f, ni = li;
    float d  = lam_re[p] * lam_re[p] + lam_im[p] * lam_im[p];
    cor[p] = (nr * lam_re[p] + ni * lam_im[p]) / d;
    coi[p] = (ni * lam_re[p] - nr * lam_im[p]) / d;
}

// Bu[m,p] = coef[p] * (fx[m,:] . (B_re[p,:] + i B_im[p,:]))
__global__ void s5_bu(const float* __restrict__ fx, const float* __restrict__ Bre,
                      const float* __restrict__ Bim, const float* __restrict__ cor,
                      const float* __restrict__ coi,
                      float* __restrict__ bur, float* __restrict__ bui)
{
    int i = blockIdx.x * blockDim.x + threadIdx.x;   // m*512+p
    if (i >= IMGS * 512) return;
    int p = i % 512, m = i / 512;
    const float* u  = fx  + (size_t)m * 512;
    const float* br = Bre + (size_t)p * 512;
    const float* bi = Bim + (size_t)p * 512;
    float sr = 0.0f, si = 0.0f;
    for (int h = 0; h < 512; ++h) { float uu = u[h]; sr += uu * br[h]; si += uu * bi[h]; }
    bur[i] = cor[p] * sr - coi[p] * si;
    bui[i] = cor[p] * si + coi[p] * sr;
}

// sequential scan over L=16: x_k = Lb * x_{k-1} + Bu_k  (per batch, per state)
__global__ void s5_scan(const float* __restrict__ Lbr, const float* __restrict__ Lbi,
                        const float* __restrict__ bur, const float* __restrict__ bui,
                        float* __restrict__ xsr, float* __restrict__ xsi)
{
    int i = blockIdx.x * blockDim.x + threadIdx.x;   // b*512+p
    if (i >= 4 * 512) return;
    int p = i % 512, bb = i / 512;
    float lr = Lbr[p], li = Lbi[p];
    float xr = 0.0f, xi = 0.0f;
    for (int l = 0; l < 16; ++l) {
        size_t o = (size_t)(bb * 16 + l) * 512 + p;
        float nr = lr * xr - li * xi + bur[o];
        float ni = lr * xi + li * xr + bui[o];
        xr = nr; xi = ni;
        xsr[o] = xr; xsi[o] = xi;
    }
}

// y = 2*Re(xs . C^T) + u*D ; hout = gelu(y) + u
__global__ void s5_out(const float* __restrict__ xsr, const float* __restrict__ xsi,
                       const float* __restrict__ Cre, const float* __restrict__ Cim,
                       const float* __restrict__ D, const float* __restrict__ fx,
                       float* __restrict__ hout)
{
    int i = blockIdx.x * blockDim.x + threadIdx.x;   // m*512+h
    if (i >= IMGS * 512) return;
    int h = i % 512, m = i / 512;
    const float* xr = xsr + (size_t)m * 512;
    const float* xi = xsi + (size_t)m * 512;
    const float* cr = Cre + (size_t)h * 512;
    const float* ci = Cim + (size_t)h * 512;
    float acc = 0.0f;
    for (int p = 0; p < 512; ++p) acc += xr[p] * cr[p] - xi[p] * ci[p];
    float u = fx[i];
    float y = 2.0f * acc + u * D[h];
    hout[i] = gelu_erf(y) + u;
}

// GEGLU gate: t[m,h] = z[m,h] * gelu(z[m,512+h]),  z is [64,1024]
__global__ void geglu_kernel(const float* __restrict__ z, float* __restrict__ t)
{
    int i = blockIdx.x * blockDim.x + threadIdx.x;
    if (i >= IMGS * 512) return;
    int h = i % 512, m = i / 512;
    float a = z[(size_t)m * 1024 + h];
    float g = z[(size_t)m * 1024 + 512 + h];
    t[i] = a * gelu_erf(g);
}

// ---------------------------------------------------------------------------
// Host driver
// ---------------------------------------------------------------------------
extern "C" void kernel_launch(void* const* d_in, const int* in_sizes, int n_in,
                              void* d_out, int out_size, void* d_ws, size_t ws_size,
                              hipStream_t stream)
{
    (void)in_sizes; (void)n_in; (void)out_size; (void)ws_size;
    const float* x = (const float*)d_in[0];
    auto F = [&](int i) { return (const float*)d_in[i]; };

    // ---- workspace carve ----
    char* base = (char*)d_ws;
    size_t off = 0;
    auto alloc = [&](size_t bytes) -> void* {
        void* p = base + off;
        off = (off + bytes + 255) & ~(size_t)255;
        return p;
    };
    const size_t SBIG = (size_t)IMGS * NOUT * OHW * OHW;  // 6,422,528 floats
    float* bufs[4];
    for (int i = 0; i < 4; ++i) bufs[i] = (float*)alloc(SBIG * 4);
    _Float16* wpack = (_Float16*)alloc((size_t)NOUT * 4608 * sizeof(_Float16));

    float* pooled = (float*)alloc(IMGS * 512 * 4);
    float* lnb    = (float*)alloc(IMGS * 512 * 4);
    float* hseq   = (float*)alloc(IMGS * 512 * 4);
    float* fx1    = (float*)alloc(IMGS * 512 * 4);
    float* fx2    = (float*)alloc(IMGS * 512 * 4);
    float* h2     = (float*)alloc(IMGS * 512 * 4);
    float* h3     = (float*)alloc(IMGS * 512 * 4);
    float* tbuf   = (float*)alloc(IMGS * 512 * 4);
    float* zbuf   = (float*)alloc(IMGS * 1024 * 4);
    float* bur    = (float*)alloc(IMGS * 512 * 4);
    float* bui    = (float*)alloc(IMGS * 512 * 4);
    float* xsr    = (float*)alloc(IMGS * 512 * 4);
    float* xsi    = (float*)alloc(IMGS * 512 * 4);
    float* p1     = (float*)alloc(IMGS * 256 * 4);
    float* Lbr = (float*)alloc(512 * 4);
    float* Lbi = (float*)alloc(512 * 4);
    float* cor = (float*)alloc(512 * 4);
    float* coi = (float*)alloc(512 * 4);

    // ---- block layout in d_in ----
    const int ci_[6]   = {3, 6, 12, 24, 48, 512};
    const int co_[6]   = {6, 12, 24, 48, 512, 512};
    const int st_[6]   = {2, 2, 2, 2, 1, 1};
    const int hin_[6]  = {224, 112, 56, 28, 14, 14};
    const int hout_[6] = {112, 56, 28, 14, 14, 14};
    const int binb[6]  = {1, 16, 31, 46, 61, 76};  // blocks 0-4: 15 arrays, block 5: 10

    const float* hcur = x;

    // ---- blocks 0..3 : direct conv path ----
    for (int bk = 0; bk < 4; ++bk) {
        int ci = ci_[bk], co = co_[bk], s = st_[bk];
        int Hin = hin_[bk], Ho = hout_[bk];
        int bi = binb[bk];
        float* free3[3]; int c = 0;
        for (int q = 0; q < 4; ++q)
            if ((const float*)bufs[q] != hcur && c < 3) free3[c++] = bufs[q];
        float* idn = free3[0]; float* t1 = free3[1]; float* t2 = free3[2];

        int total = IMGS * co * Ho * Ho;
        int nb = (total + 255) / 256;
        conv3_bn_kernel<<<nb, 256, 0, stream>>>(hcur, F(bi + 10),
            F(bi + 11), F(bi + 12), F(bi + 13), F(bi + 14),
            nullptr, idn, ci, co, Hin, Hin, Ho, Ho, s, 0, 1, 0, total);
        conv3_bn_kernel<<<nb, 256, 0, stream>>>(hcur, F(bi + 0),
            F(bi + 1), F(bi + 2), F(bi + 3), F(bi + 4),
            nullptr, t1, ci, co, Hin, Hin, Ho, Ho, s, (s == 2 ? 0 : 1), 3, 1, total);
        conv3_bn_kernel<<<nb, 256, 0, stream>>>(t1, F(bi + 5),
            F(bi + 6), F(bi + 7), F(bi + 8), F(bi + 9),
            idn, t2, co, co, Ho, Ho, Ho, Ho, 1, 1, 3, 1, total);
        hcur = t2;
    }

    dim3 gg(MPOS / BM, NOUT / BN);  // 98 x 4

    // ---- block 4 : 48 -> 512 (WMMA implicit GEMM, TDM weight streaming) ----
    {
        int bi = binb[4];
        float* free3[3]; int c = 0;
        for (int q = 0; q < 4; ++q)
            if ((const float*)bufs[q] != hcur && c < 3) free3[c++] = bufs[q];
        float* idn = free3[0]; float* t1 = free3[1]; float* t2 = free3[2];

        // dconv 1x1 (K=48 pad to 64)
        pack_w_f16<<<(512 * 64 + 255) / 256, 256, 0, stream>>>(F(bi + 10), wpack, 48, 1, 64, 512);
        conv_igemm_wmma<48, 1><<<gg, 256, 0, stream>>>(hcur, wpack, idn, nullptr,
            F(bi + 11), F(bi + 12), F(bi + 13), F(bi + 14), 64, 0);
        // conv1 (K=432 pad to 448) + relu
        pack_w_f16<<<(512 * 448 + 255) / 256, 256, 0, stream>>>(F(bi + 0), wpack, 48, 3, 448, 512);
        conv_igemm_wmma<48, 3><<<gg, 256, 0, stream>>>(hcur, wpack, t1, nullptr,
            F(bi + 1), F(bi + 2), F(bi + 3), F(bi + 4), 448, 1);
        // conv2 (K=4608) + idn + relu
        pack_w_f16<<<(512 * 4608 + 255) / 256, 256, 0, stream>>>(F(bi + 5), wpack, 512, 3, 4608, 512);
        conv_igemm_wmma<512, 3><<<gg, 256, 0, stream>>>(t1, wpack, t2, idn,
            F(bi + 6), F(bi + 7), F(bi + 8), F(bi + 9), 4608, 1);
        hcur = t2;
    }

    // ---- block 5 : 512 -> 512 (identity residual) ----
    {
        int bi = binb[5];
        float* free3[3]; int c = 0;
        for (int q = 0; q < 4; ++q)
            if ((const float*)bufs[q] != hcur && c < 3) free3[c++] = bufs[q];
        float* t1 = free3[0]; float* t2 = free3[1];

        pack_w_f16<<<(512 * 4608 + 255) / 256, 256, 0, stream>>>(F(bi + 0), wpack, 512, 3, 4608, 512);
        conv_igemm_wmma<512, 3><<<gg, 256, 0, stream>>>(hcur, wpack, t1, nullptr,
            F(bi + 1), F(bi + 2), F(bi + 3), F(bi + 4), 4608, 1);
        pack_w_f16<<<(512 * 4608 + 255) / 256, 256, 0, stream>>>(F(bi + 5), wpack, 512, 3, 4608, 512);
        conv_igemm_wmma<512, 3><<<gg, 256, 0, stream>>>(t1, wpack, t2, hcur,
            F(bi + 6), F(bi + 7), F(bi + 8), F(bi + 9), 4608, 1);
        hcur = t2;
    }

    // ---- pool + bridge ----
    pool_kernel<<<(IMGS * 512 + 255) / 256, 256, 0, stream>>>(hcur, pooled);
    ln_kernel<<<IMGS, 256, 0, stream>>>(pooled, F(86), F(87), lnb, 512);
    gemm_small<<<(IMGS * 512 + 255) / 256, 256, 0, stream>>>(lnb, F(88), F(89), nullptr,
                                                             hseq, IMGS, 512, 512, 0);

    // ---- S5 block ----
    ln_kernel<<<IMGS, 256, 0, stream>>>(hseq, F(90), F(91), fx1, 512);
    s5_coeff<<<2, 256, 0, stream>>>(F(92), F(93), F(99), Lbr, Lbi, cor, coi);
    s5_bu<<<(IMGS * 512 + 255) / 256, 256, 0, stream>>>(fx1, F(94), F(95), cor, coi, bur, bui);
    s5_scan<<<8, 256, 0, stream>>>(Lbr, Lbi, bur, bui, xsr, xsi);
    s5_out<<<(IMGS * 512 + 255) / 256, 256, 0, stream>>>(xsr, xsi, F(96), F(97), F(98), fx1, h2);

    // ---- GEGLU FFN ----
    ln_kernel<<<IMGS, 256, 0, stream>>>(h2, F(100), F(101), fx2, 512);
    gemm_small<<<(IMGS * 1024 + 255) / 256, 256, 0, stream>>>(fx2, F(102), nullptr, nullptr,
                                                              zbuf, IMGS, 1024, 512, 0);
    geglu_kernel<<<(IMGS * 512 + 255) / 256, 256, 0, stream>>>(zbuf, tbuf);
    gemm_small<<<(IMGS * 512 + 255) / 256, 256, 0, stream>>>(tbuf, F(103), nullptr, fx2,
                                                             h3, IMGS, 512, 512, 0);

    // ---- head ----
    gemm_small<<<(IMGS * 256 + 255) / 256, 256, 0, stream>>>(h3, F(104), F(105), nullptr,
                                                             p1, IMGS, 256, 512, 1);
    gemm_small<<<(IMGS * 400 + 255) / 256, 256, 0, stream>>>(p1, F(106), F(107), nullptr,
                                                             (float*)d_out, IMGS, 400, 256, 0);
}
